// CausalSelfAttention_33354716020839
// MI455X (gfx1250) — matrix operand order repaired and checked by reading, not compiled, not run on hardware
//
#include <hip/hip_runtime.h>

// ---------------------------------------------------------------------------
// CDNA5 (gfx1250) causal self-attention, bf16 WMMA pipeline.
// Software-pipelined global->LDS staging with LDS double buffering;
// all LDS fragment traffic as ds_load_b128.
//   B=4, T=2048, C=1024, H=16, DH=64
// ---------------------------------------------------------------------------

typedef __bf16 bf16_t;
typedef __attribute__((ext_vector_type(16))) __bf16 v16bf;
typedef __attribute__((ext_vector_type(8)))  float  v8f;

#define B_  4
#define T_  2048
#define C_  1024
#define H_  16
#define DH_ 64
#define BT_ (B_ * T_)

// -------------------- WMMA fragment helpers (wave32) -----------------------
// Fragment layouts as used here (A and B share a contiguous k order):
//  A (16x32 bf16): lane half h = lane>>4, row m = lane&15.
//    elements 0..7  <-> k = h*16     .. h*16+7  (contiguous, 16B)
//    elements 8..15 <-> k = h*16+8   .. +7      (contiguous, 16B)
//  B (32x16 bf16): lanes 0-15 hold K=0..15, lanes 16-31 hold K=16..31;
//    N = lane&15.  With an [n][k] LDS tile this is 16 contiguous bf16/lane.
//  C/D (16x16 f32): 8 VGPRs; slot r -> row m = h*8 + r, col n = lane&15

union frag_cast { uint4 u[2]; v16bf v; };

__device__ inline v8f v8f_zero() {
  v8f z;
#pragma unroll
  for (int i = 0; i < 8; ++i) z[i] = 0.0f;
  return z;
}

__device__ inline v16bf load_A16x32(const bf16_t* __restrict__ p, int ld, int lane) {
  const int half = lane >> 4;
  const int m    = lane & 15;
  const bf16_t* base = p + m * ld + half * 16;
  frag_cast t;
  t.u[0] = *reinterpret_cast<const uint4*>(base);       // k = h*16 .. +7
  t.u[1] = *reinterpret_cast<const uint4*>(base + 8);   // k = h*16+8 .. +15
  return t.v;
}

__device__ inline v16bf load_B32x16_cm(const bf16_t* __restrict__ p, int ld, int lane) {
  const int half = lane >> 4;
  const int n    = lane & 15;
  const bf16_t* base = p + n * ld + half * 16;
  frag_cast t;
  t.u[0] = *reinterpret_cast<const uint4*>(base);      // k = h*16 .. +7
  t.u[1] = *reinterpret_cast<const uint4*>(base + 8);  // k = h*16+8 .. +15
  return t.v;
}

__device__ inline v8f wmma_bf16(v16bf a, v16bf b, v8f c) {
  return __builtin_amdgcn_wmma_f32_16x16x32_bf16(false, a, false, b, (short)0, c,
                                                 false, false);
}

// ---------------- two-phase staging helpers (8 x 16B chunks) ---------------
__device__ inline void gload8(uint4 r[8], const bf16_t* __restrict__ pa,
                              const bf16_t* __restrict__ pb) {
  const uint4* sa = reinterpret_cast<const uint4*>(pa);
  const uint4* sb = reinterpret_cast<const uint4*>(pb);
#pragma unroll
  for (int u = 0; u < 4; ++u) { r[u] = sa[u]; r[4 + u] = sb[u]; }
}
__device__ inline void sstore8(bf16_t* da, bf16_t* db, const uint4 r[8]) {
  uint4* xa = reinterpret_cast<uint4*>(da);
  uint4* xb = reinterpret_cast<uint4*>(db);
#pragma unroll
  for (int u = 0; u < 4; ++u) { xa[u] = r[u]; xb[u] = r[4 + u]; }
}

// -------------------------- fp32 -> bf16 convert ---------------------------
__global__ __launch_bounds__(256) void f32_to_bf16_kernel(
    const float* __restrict__ in, bf16_t* __restrict__ out, int n4) {
  int i = blockIdx.x * 256 + threadIdx.x;
  if (i < n4) {
    float4 f = reinterpret_cast<const float4*>(in)[i];
    out[i * 4 + 0] = (bf16_t)f.x;
    out[i * 4 + 1] = (bf16_t)f.y;
    out[i * 4 + 2] = (bf16_t)f.z;
    out[i * 4 + 3] = (bf16_t)f.w;
  }
}

// ------------------- fp32 -> bf16 transpose (32x32 tiles) ------------------
__global__ __launch_bounds__(256) void transpose_f32_bf16_kernel(
    const float* __restrict__ in, bf16_t* __restrict__ out, int R, int Cc) {
  __shared__ float tile[32][33];
  const int bx = blockIdx.x % (Cc / 32);
  const int by = blockIdx.x / (Cc / 32);
  const int tx = threadIdx.x & 31;
  const int ty = threadIdx.x >> 5;
#pragma unroll
  for (int yy = ty; yy < 32; yy += 8)
    tile[yy][tx] = in[(size_t)(by * 32 + yy) * Cc + bx * 32 + tx];
  __syncthreads();
#pragma unroll
  for (int yy = ty; yy < 32; yy += 8)
    out[(size_t)(bx * 32 + yy) * R + by * 32 + tx] = (bf16_t)tile[tx][yy];
}

// ------------------------------ GEMM core ----------------------------------
#define G_TM 128
#define G_TN 128
#define G_TK 32
#define G_LD 40     // 32 + 8 pad (80B rows, 16B aligned)

__device__ inline void gemm_step(const bf16_t* __restrict__ sAbuf,
                                 const bf16_t* __restrict__ sBbuf,
                                 int wm, int wn, int lane, v8f (&acc)[4][4]) {
  v16bf af[4], bfm[4];
#pragma unroll
  for (int i = 0; i < 4; ++i)
    af[i] = load_A16x32(sAbuf + (wm * 64 + i * 16) * G_LD, G_LD, lane);
#pragma unroll
  for (int j = 0; j < 4; ++j)
    bfm[j] = load_B32x16_cm(sBbuf + (wn * 64 + j * 16) * G_LD, G_LD, lane);
#pragma unroll
  for (int i = 0; i < 4; ++i)
#pragma unroll
    for (int j = 0; j < 4; ++j)
      acc[i][j] = wmma_bf16(af[i], bfm[j], acc[i][j]);
}

// ------------------------------ QKV GEMM -----------------------------------
// X[8192,1024] bf16 @ W (as WT[3072,1024] bf16) -> scatter Q/K/Vt (bf16).
__global__ __launch_bounds__(128) void qkv_gemm_kernel(
    const bf16_t* __restrict__ X, const bf16_t* __restrict__ WT,
    bf16_t* __restrict__ Qo, bf16_t* __restrict__ Ko, bf16_t* __restrict__ Vt) {
  __shared__ __align__(16) bf16_t sA0[G_TM * G_LD];
  __shared__ __align__(16) bf16_t sB0[G_TN * G_LD];
  __shared__ __align__(16) bf16_t sA1[G_TM * G_LD];
  __shared__ __align__(16) bf16_t sB1[G_TN * G_LD];

  const int tid  = threadIdx.x;
  const int lane = tid & 31;
  const int wave = tid >> 5;
  const int wm   = wave >> 1;
  const int wn   = wave & 1;

  const int nTilesN = (3 * C_) / G_TN;           // 24
  const int row0 = (blockIdx.x / nTilesN) * G_TM;
  const int col0 = (blockIdx.x % nTilesN) * G_TN;

  const bf16_t* gA = X  + (size_t)(row0 + tid) * C_;
  const bf16_t* gB = WT + (size_t)(col0 + tid) * C_;
  bf16_t* lA0 = sA0 + tid * G_LD;
  bf16_t* lB0 = sB0 + tid * G_LD;
  bf16_t* lA1 = sA1 + tid * G_LD;
  bf16_t* lB1 = sB1 + tid * G_LD;

  v8f acc[4][4];
#pragma unroll
  for (int i = 0; i < 4; ++i)
#pragma unroll
    for (int j = 0; j < 4; ++j) acc[i][j] = v8f_zero();

  uint4 st[8];
  gload8(st, gA, gB);
  sstore8(lA0, lB0, st);
  __syncthreads();

  for (int k0 = 0; k0 < C_; k0 += 2 * G_TK) {
    // ---- half 1: compute buf0, stage k0+32 -> buf1 ----
    gload8(st, gA + k0 + G_TK, gB + k0 + G_TK);
    if (k0 + 3 * G_TK < C_) {
      __builtin_prefetch(gA + k0 + 3 * G_TK, 0, 1);
      __builtin_prefetch(gB + k0 + 3 * G_TK, 0, 1);
    }
    gemm_step(sA0, sB0, wm, wn, lane, acc);
    sstore8(lA1, lB1, st);
    __syncthreads();
    // ---- half 2: compute buf1, stage k0+64 -> buf0 ----
    const bool more = (k0 + 2 * G_TK < C_);
    if (more) gload8(st, gA + k0 + 2 * G_TK, gB + k0 + 2 * G_TK);
    gemm_step(sA1, sB1, wm, wn, lane, acc);
    if (more) sstore8(lA0, lB0, st);
    __syncthreads();
  }

  // Scatter epilogue: Q,K natural [B,H,T,DH]; V transposed [B,H,DH,T].
  const int half = lane >> 4;
  const int nn   = lane & 15;
#pragma unroll
  for (int i = 0; i < 4; ++i) {
#pragma unroll
    for (int j = 0; j < 4; ++j) {
#pragma unroll
      for (int r = 0; r < 8; ++r) {
        const int grow = row0 + wm * 64 + i * 16 + (half << 3) + r;
        const int gcol = col0 + wn * 64 + j * 16 + nn;
        const bf16_t bv = (bf16_t)acc[i][j][r];
        const int which = gcol >> 10;
        const int cc    = gcol & 1023;
        const int h     = cc >> 6;
        const int d     = cc & 63;
        const int b     = grow >> 11;
        const int t     = grow & 2047;
        const size_t bh = (size_t)b * H_ + h;
        if (which == 0)      Qo[(bh * T_ + t) * DH_ + d] = bv;
        else if (which == 1) Ko[(bh * T_ + t) * DH_ + d] = bv;
        else                 Vt[(bh * DH_ + d) * T_ + t] = bv;
      }
    }
  }
}

// --------------------------- Flash attention -------------------------------
#define QB 64
#define KB 64
#define F_LD 72     // 64 + 8 pad (144B rows, 16B aligned)

__global__ __launch_bounds__(128) void flash_attn_kernel(
    const bf16_t* __restrict__ Q, const bf16_t* __restrict__ K,
    const bf16_t* __restrict__ Vt, bf16_t* __restrict__ Y) {
  __shared__ __align__(16) bf16_t sQ[QB][F_LD];        // [t_q][d]
  __shared__ __align__(16) bf16_t sK[2][KB][F_LD];     // [t_k][d]
  __shared__ __align__(16) bf16_t sVt[2][DH_][F_LD];   // [d][t_k]
  __shared__ __align__(16) bf16_t sP[4][16][F_LD];     // per-wave P [m][t_k]

  const int tid  = threadIdx.x;
  const int lane = tid & 31;
  const int wave = tid >> 5;
  const int half = lane >> 4;
  const int nn   = lane & 15;

  const int qb = blockIdx.x % (T_ / QB);
  const int bh = blockIdx.x / (T_ / QB);
  const int b  = bh / H_;
  const int h  = bh % H_;
  const int qbase = qb * QB;
  const size_t baseQV = (size_t)bh * T_ * DH_;
  const size_t baseVt = (size_t)bh * DH_ * T_;

  const int r    = tid >> 1;
  const int cseg = (tid & 1) * 32;

  // Prologue: stage Q tile and KV block 0 into buffer 0
  {
    uint4 st[8];
    gload8(st, Q + baseQV + (size_t)(qbase + r) * DH_ + cseg,
               K + baseQV + (size_t)r * DH_ + cseg);      // kbase=0
    sstore8(&sQ[r][cseg], &sK[0][r][cseg], st);
    uint4 sv[4];
    const uint4* src = reinterpret_cast<const uint4*>(
        Vt + baseVt + (size_t)r * T_ + cseg);             // kbase=0
#pragma unroll
    for (int u = 0; u < 4; ++u) sv[u] = src[u];
    uint4* dst = reinterpret_cast<uint4*>(&sVt[0][r][cseg]);
#pragma unroll
    for (int u = 0; u < 4; ++u) dst[u] = sv[u];
  }
  __syncthreads();

  v16bf qf[2];
  qf[0] = load_A16x32(&sQ[wave * 16][0],  F_LD, lane);
  qf[1] = load_A16x32(&sQ[wave * 16][32], F_LD, lane);

  float mrow[8], lrow[8];
  v8f o[4];
#pragma unroll
  for (int rr = 0; rr < 8; ++rr) { mrow[rr] = -3.0e38f; lrow[rr] = 0.0f; }
#pragma unroll
  for (int nt = 0; nt < 4; ++nt) o[nt] = v8f_zero();

  const float scale = 0.125f;  // 1/sqrt(64)

  for (int j = 0; j <= qb; ++j) {
    const int cur = j & 1;
    const int nxt = cur ^ 1;
    const int kbase = j * KB;
    const bool more = (j < qb);

    // Issue next block's global loads early (hidden behind S/softmax)
    uint4 stk[8];
    if (more) {
      const int kb2 = kbase + KB;
      gload8(stk, K + baseQV + (size_t)(kb2 + r) * DH_ + cseg,
                  Vt + baseVt + (size_t)r * T_ + kb2 + cseg);
    }

    // S = Q * K^T  (16x64 per wave)
    v8f s[4];
#pragma unroll
    for (int nt = 0; nt < 4; ++nt) {
      v8f c = v8f_zero();
      v16bf b0 = load_B32x16_cm(&sK[cur][nt * 16][0],  F_LD, lane);
      v16bf b1 = load_B32x16_cm(&sK[cur][nt * 16][32], F_LD, lane);
      c = wmma_bf16(qf[0], b0, c);
      c = wmma_bf16(qf[1], b1, c);
      s[nt] = c;
    }

    // scale + causal mask (diagonal block only)
    const bool diag = (j == qb);
#pragma unroll
    for (int nt = 0; nt < 4; ++nt) {
#pragma unroll
      for (int rr = 0; rr < 8; ++rr) {
        float v = s[nt][rr] * scale;
        if (diag) {
          const int tk = kbase + nt * 16 + nn;
          const int tq = qbase + wave * 16 + (half << 3) + rr;
          if (tk > tq) v = -1.0e30f;
        }
        s[nt][rr] = v;
      }
    }

    // online softmax (row m = half*8+rr in slot rr across 16 lanes)
    float rmax[8];
#pragma unroll
    for (int rr = 0; rr < 8; ++rr) {
      float mv = s[0][rr];
#pragma unroll
      for (int nt = 1; nt < 4; ++nt) mv = fmaxf(mv, s[nt][rr]);
#pragma unroll
      for (int off = 1; off < 16; off <<= 1)
        mv = fmaxf(mv, __shfl_xor(mv, off, 16));
      rmax[rr] = mv;
    }

    float factor[8], rsum[8];
#pragma unroll
    for (int rr = 0; rr < 8; ++rr) {
      const float mnew = fmaxf(mrow[rr], rmax[rr]);
      factor[rr] = __expf(mrow[rr] - mnew);
      mrow[rr] = mnew;
      rsum[rr] = 0.0f;
    }

#pragma unroll
    for (int nt = 0; nt < 4; ++nt) {
#pragma unroll
      for (int rr = 0; rr < 8; ++rr) {
        const float p = __expf(s[nt][rr] - mrow[rr]);
        rsum[rr] += p;
        sP[wave][(half << 3) + rr][nt * 16 + nn] = (bf16_t)p;
      }
    }
#pragma unroll
    for (int rr = 0; rr < 8; ++rr) {
      float sm = rsum[rr];
#pragma unroll
      for (int off = 1; off < 16; off <<= 1) sm += __shfl_xor(sm, off, 16);
      lrow[rr] = lrow[rr] * factor[rr] + sm;
    }
#pragma unroll
    for (int nt = 0; nt < 4; ++nt)
#pragma unroll
      for (int rr = 0; rr < 8; ++rr) o[nt][rr] *= factor[rr];

    // stage next KV block into the buffer whose last reads were before the
    // previous iteration's end barrier
    if (more) sstore8(&sK[nxt][r][cseg], &sVt[nxt][r][cseg], stk);

    __syncthreads();  // publish sP + complete staging of nxt

    // O += P * V (from sP and sVt[cur])
    v16bf pf0 = load_A16x32(&sP[wave][0][0],  F_LD, lane);
    v16bf pf1 = load_A16x32(&sP[wave][0][32], F_LD, lane);
#pragma unroll
    for (int nt = 0; nt < 4; ++nt) {
      v16bf vb0 = load_B32x16_cm(&sVt[cur][nt * 16][0],  F_LD, lane);
      v16bf vb1 = load_B32x16_cm(&sVt[cur][nt * 16][32], F_LD, lane);
      o[nt] = wmma_bf16(pf0, vb0, o[nt]);
      o[nt] = wmma_bf16(pf1, vb1, o[nt]);
    }

    __syncthreads();  // cur reads done before it is restaged in j+1
  }

  // Normalize and store to Y[b, t, h*64 + d] (bf16)
#pragma unroll
  for (int nt = 0; nt < 4; ++nt) {
#pragma unroll
    for (int rr = 0; rr < 8; ++rr) {
      const int t = qbase + wave * 16 + (half << 3) + rr;
      const int d = nt * 16 + nn;
      const float val = o[nt][rr] / lrow[rr];
      Y[((size_t)b * T_ + t) * C_ + h * DH_ + d] = (bf16_t)val;
    }
  }
}

// --------------------------- Output projection -----------------------------
__global__ __launch_bounds__(128) void proj_gemm_kernel(
    const bf16_t* __restrict__ Yb, const bf16_t* __restrict__ WpT,
    float* __restrict__ Out) {
  __shared__ __align__(16) bf16_t sA0[G_TM * G_LD];
  __shared__ __align__(16) bf16_t sB0[G_TN * G_LD];
  __shared__ __align__(16) bf16_t sA1[G_TM * G_LD];
  __shared__ __align__(16) bf16_t sB1[G_TN * G_LD];

  const int tid  = threadIdx.x;
  const int lane = tid & 31;
  const int wave = tid >> 5;
  const int wm   = wave >> 1;
  const int wn   = wave & 1;

  const int nTilesN = C_ / G_TN;                 // 8
  const int row0 = (blockIdx.x / nTilesN) * G_TM;
  const int col0 = (blockIdx.x % nTilesN) * G_TN;

  const bf16_t* gA = Yb  + (size_t)(row0 + tid) * C_;
  const bf16_t* gB = WpT + (size_t)(col0 + tid) * C_;
  bf16_t* lA0 = sA0 + tid * G_LD;
  bf16_t* lB0 = sB0 + tid * G_LD;
  bf16_t* lA1 = sA1 + tid * G_LD;
  bf16_t* lB1 = sB1 + tid * G_LD;

  v8f acc[4][4];
#pragma unroll
  for (int i = 0; i < 4; ++i)
#pragma unroll
    for (int j = 0; j < 4; ++j) acc[i][j] = v8f_zero();

  uint4 st[8];
  gload8(st, gA, gB);
  sstore8(lA0, lB0, st);
  __syncthreads();

  for (int k0 = 0; k0 < C_; k0 += 2 * G_TK) {
    gload8(st, gA + k0 + G_TK, gB + k0 + G_TK);
    if (k0 + 3 * G_TK < C_) {
      __builtin_prefetch(gA + k0 + 3 * G_TK, 0, 1);
      __builtin_prefetch(gB + k0 + 3 * G_TK, 0, 1);
    }
    gemm_step(sA0, sB0, wm, wn, lane, acc);
    sstore8(lA1, lB1, st);
    __syncthreads();
    const bool more = (k0 + 2 * G_TK < C_);
    if (more) gload8(st, gA + k0 + 2 * G_TK, gB + k0 + 2 * G_TK);
    gemm_step(sA1, sB1, wm, wn, lane, acc);
    if (more) sstore8(lA0, lB0, st);
    __syncthreads();
  }

  const int half = lane >> 4;
  const int nn   = lane & 15;
#pragma unroll
  for (int i = 0; i < 4; ++i)
#pragma unroll
    for (int j = 0; j < 4; ++j)
#pragma unroll
      for (int rr = 0; rr < 8; ++rr) {
        const int grow = row0 + wm * 64 + i * 16 + (half << 3) + rr;
        const int gcol = col0 + wn * 64 + j * 16 + nn;
        Out[(size_t)grow * C_ + gcol] = acc[i][j][rr];
      }
}

// ------------------------------ launcher -----------------------------------
extern "C" void kernel_launch(void* const* d_in, const int* in_sizes, int n_in,
                              void* d_out, int out_size, void* d_ws, size_t ws_size,
                              hipStream_t stream) {
  const float* x      = (const float*)d_in[0];   // [B,T,C]
  const float* w_attn = (const float*)d_in[1];   // [C,3C]
  const float* w_proj = (const float*)d_in[2];   // [C,C]
  float* out = (float*)d_out;                    // [B,T,C]

  char* ws = (char*)d_ws;
  size_t off = 0;
  auto alloc = [&](size_t bytes) -> void* {
    void* p = ws + off;
    off += (bytes + 255) & ~(size_t)255;
    return p;
  };

  bf16_t* xb  = (bf16_t*)alloc((size_t)BT_ * C_ * 2);       // [BT, C]
  bf16_t* waT = (bf16_t*)alloc((size_t)(3 * C_) * C_ * 2);  // [3C, C]
  bf16_t* wpT = (bf16_t*)alloc((size_t)C_ * C_ * 2);        // [C, C]
  bf16_t* Qb  = (bf16_t*)alloc((size_t)BT_ * C_ * 2);       // [B,H,T,DH]
  bf16_t* Kb  = (bf16_t*)alloc((size_t)BT_ * C_ * 2);       // [B,H,T,DH]
  bf16_t* Vtb = (bf16_t*)alloc((size_t)BT_ * C_ * 2);       // [B,H,DH,T]
  bf16_t* Yb  = (bf16_t*)alloc((size_t)BT_ * C_ * 2);       // [B,T,C]

  {
    int n4 = (BT_ * C_) / 4;
    f32_to_bf16_kernel<<<(n4 + 255) / 256, 256, 0, stream>>>(x, xb, n4);
    transpose_f32_bf16_kernel<<<(3 * C_ / 32) * (C_ / 32), 256, 0, stream>>>(
        w_attn, waT, C_, 3 * C_);
    transpose_f32_bf16_kernel<<<(C_ / 32) * (C_ / 32), 256, 0, stream>>>(
        w_proj, wpT, C_, C_);
  }

  qkv_gemm_kernel<<<(BT_ / G_TM) * ((3 * C_) / G_TN), 128, 0, stream>>>(
      xb, waT, Qb, Kb, Vtb);

  flash_attn_kernel<<<(T_ / QB) * B_ * H_, 128, 0, stream>>>(Qb, Kb, Vtb, Yb);

  proj_gemm_kernel<<<(BT_ / G_TM) * (C_ / G_TN), 128, 0, stream>>>(Yb, wpT, out);
}
